// LightningAttn3Attention_59657095741555
// MI455X (gfx1250) — compile-verified, matrix operand order checked by reading, not run on hardware
//
#include <hip/hip_runtime.h>
#include <hip/hip_bf16.h>
#include <math.h>

// ---------------------------------------------------------------------------
// LightningAttn3 on gfx1250: one-shot fp32->bf16 cast pass, then bf16 WMMA
// GEMMs fed by GLOBAL_LOAD_ASYNC_TO_LDS_B128, then fused-RoPE chunked linear
// attention (WMMA, in-place decay scaling), then WMMA output projection.
// ---------------------------------------------------------------------------

typedef __attribute__((ext_vector_type(16))) __bf16 v16bf;
typedef __attribute__((ext_vector_type(8)))  float  v8f;

#define B_      2
#define S_      4096
#define D_      2048
#define H_      16
#define HKV_    4
#define DH_     128
#define CHUNK_  64
#define NC_     (S_ / CHUNK_)
#define GROUPS_ (H_ / HKV_)
#define MROWS_  (B_ * S_)     // 8192

// native gfx1250 bf16 converts (v_cvt_*), not software rounding
__device__ __forceinline__ __bf16 f2bf(float f) { return (__bf16)f; }
__device__ __forceinline__ float  bf2f(__bf16 b) { return (float)b; }

__device__ __forceinline__ v8f wmma_bf16(v16bf a, v16bf b, v8f c) {
  return __builtin_amdgcn_wmma_f32_16x16x32_bf16(false, a, false, b,
                                                 (short)0, c, false, false);
}

// CDNA5 async copy global->LDS (ASYNCcnt), 16 bytes per lane.
__device__ __forceinline__ void async_ld_b128(void* lds_ptr, const void* gptr) {
  unsigned lds_off = (unsigned)(uintptr_t)lds_ptr;                 // LDS addr = low 32 bits
  unsigned long long ga = (unsigned long long)(uintptr_t)gptr;
  asm volatile("global_load_async_to_lds_b128 %0, %1, off"
               :: "v"(lds_off), "v"(ga) : "memory");
}
__device__ __forceinline__ void wait_async0() {
  asm volatile("s_wait_asynccnt 0" ::: "memory");
}

// A-fragment (16x32 bf16, ISA 7.12.2): lane&15 = row; lanes 0-15 carry K
// {kk..+7, kk+16..+23}, lanes 16-31 carry K {kk+8..+15, kk+24..+31}.
__device__ __forceinline__ v16bf frag_row(const __bf16* base, int ld, int lane, int kk) {
  const __bf16* row = base + (lane & 15) * ld;
  const int kb = kk + ((lane < 16) ? 0 : 8);
  v16bf f;
#pragma unroll
  for (int t = 0; t < 8; ++t) { f[t] = row[kb + t]; f[t + 8] = row[kb + 16 + t]; }
  return f;
}
// B-fragment for A@B with B row-major [K x N]: lane&15 = column.
__device__ __forceinline__ v16bf frag_col(const __bf16* base, int ld, int lane, int kk) {
  const __bf16* col = base + (lane & 15);
  const int kb = kk + ((lane < 16) ? 0 : 8);
  v16bf f;
#pragma unroll
  for (int t = 0; t < 8; ++t) { f[t] = col[(kb + t) * ld]; f[t + 8] = col[(kb + 16 + t) * ld]; }
  return f;
}

__device__ __forceinline__ void store_elem(float* p, float v)  { *p = v; }
__device__ __forceinline__ void store_elem(__bf16* p, float v) { *p = f2bf(v); }

// ---------------------------------------------------------------------------
// One-shot fp32 -> bf16 cast (bandwidth-bound pre-pass).
// ---------------------------------------------------------------------------
__global__ __launch_bounds__(256)
void cast_bf16_kernel(const float* __restrict__ src, __bf16* __restrict__ dst, int n) {
  for (int i = blockIdx.x * 256 + threadIdx.x; i < n; i += gridDim.x * 256)
    dst[i] = f2bf(src[i]);
}

// ---------------------------------------------------------------------------
// GEMM: C[M,N] = A[M,K] * W[N,K]^T, A/W bf16 in memory, fp32 accumulate.
// Block tile 128x64, 8 wave32s; each wave owns 16 rows x 64 cols:
// 1 A-fragment reused over 4 B-fragments -> 4 WMMAs per K=32 step.
// LDS tiles filled with GLOBAL_LOAD_ASYNC_TO_LDS_B128.
// ---------------------------------------------------------------------------
template <typename OutT>
__global__ __launch_bounds__(256)
void gemm_bf16_kernel(const __bf16* __restrict__ A, const __bf16* __restrict__ W,
                      OutT* __restrict__ C, int M, int N, int K) {
  __shared__ __bf16 As[128][32];   // 8 KB
  __shared__ __bf16 Ws[64][32];    // 4 KB

  const int bm   = blockIdx.y * 128;
  const int bn   = blockIdx.x * 64;
  const int tid  = threadIdx.x;
  const int wave = tid >> 5;
  const int lane = tid & 31;

  v8f acc[4] = {{}, {}, {}, {}};

  for (int k0 = 0; k0 < K; k0 += 32) {
    __syncthreads();    // previous iteration's LDS reads complete
    // A tile 128x32 bf16 = 512 x 16B copies (2/thread); W tile 64x32 = 256 (1/thread)
    {
      int idx = tid;
#pragma unroll
      for (int u = 0; u < 2; ++u, idx += 256) {
        const int r = idx >> 2, sg = (idx & 3) * 8;
        async_ld_b128(&As[r][sg], A + (size_t)(bm + r) * K + k0 + sg);
      }
      const int r = tid >> 2, sg = (tid & 3) * 8;
      async_ld_b128(&Ws[r][sg], W + (size_t)(bn + r) * K + k0 + sg);
      if (k0 + 32 < K)   // speculative prefetch of next A tile row
        __builtin_prefetch(A + (size_t)(bm + r) * K + k0 + 32, 0, 0);
    }
    wait_async0();
    __syncthreads();

    v16bf af = frag_row(&As[wave * 16][0], 32, lane, 0);
#pragma unroll
    for (int t = 0; t < 4; ++t) {
      v16bf bf = frag_row(&Ws[t * 16][0], 32, lane, 0);
      acc[t] = wmma_bf16(af, bf, acc[t]);
    }
  }

  const int mb = bm + wave * 16 + ((lane < 16) ? 0 : 8);
  const int nn = bn + (lane & 15);
#pragma unroll
  for (int t = 0; t < 4; ++t)
#pragma unroll
    for (int r = 0; r < 8; ++r)
      store_elem(&C[(size_t)(mb + r) * N + nn + t * 16], acc[t][r]);
}

// ---------------------------------------------------------------------------
// Lightning attention: one workgroup per (b,h), sequential over 64 chunks.
// RoPE fused into the chunk load; decay scaling done IN PLACE between phases
// (scores read raw q/k; o reads q*q_dec; kv-update reads k*k_dec).
// kv state (128x128 fp32) lives in a private global slab (L2 resident).
// ---------------------------------------------------------------------------
__global__ __launch_bounds__(256)
void lightning_attn_kernel(const __bf16* __restrict__ Q,   // [B*S, H*DH]
                           const __bf16* __restrict__ Kb,  // [B*S, HKV*DH]
                           const __bf16* __restrict__ Vb,  // [B*S, HKV*DH]
                           __bf16* __restrict__ O,         // [B*S, H*DH]
                           float* __restrict__ KV)         // [B*H, DH, DH]
{
  const int bh = blockIdx.x;            // 0..31
  const int b  = bh / H_;
  const int h  = bh % H_;
  const int hk = h / GROUPS_;
  const float lam = exp2f(-8.0f * (float)(h + 1) / (float)H_);
  const float blkdec = __expf(-lam * (float)CHUNK_);

  __shared__ __bf16 q_s[CHUNK_][DH_];     // 16 KB
  __shared__ __bf16 k_s[CHUNK_][DH_];     // 16 KB
  __shared__ __bf16 v_s[CHUNK_][DH_];     // 16 KB
  __shared__ __bf16 sc_s[CHUNK_][CHUNK_]; //  8 KB
  __shared__ float  qdec_s[CHUNK_], kdec_s[CHUNK_], dec_s[CHUNK_], ifreq_s[64];

  float* kv = KV + (size_t)bh * DH_ * DH_;

  const int tid  = threadIdx.x;
  const int wave = tid >> 5;
  const int lane = tid & 31;

  for (int i = tid; i < DH_ * DH_; i += 256) kv[i] = 0.0f;
  if (tid < CHUNK_) {                       // decay + rope tables (once)
    qdec_s[tid]  = __expf(-lam * (float)(tid + 1));
    kdec_s[tid]  = __expf(-lam * (float)(CHUNK_ - 1 - tid));
    dec_s[tid]   = __expf(-lam * (float)tid);
    ifreq_s[tid] = __powf(10000.0f, -(float)tid / 64.0f);
  }
  __threadfence();
  __syncthreads();

  for (int c = 0; c < NC_; ++c) {
    // ---- phase 0: load chunk.  q,k: RoPE in fp32 -> LDS bf16.  v: async copy.
    for (int i = tid; i < (CHUNK_ * DH_) / 8; i += 256) {
      const int r = i / (DH_ / 8), sg = (i % (DH_ / 8)) * 8;
      async_ld_b128(&v_s[r][sg],
                    Vb + (size_t)(b * S_ + c * CHUNK_ + r) * (HKV_ * DH_) + hk * DH_ + sg);
    }
    for (int i = tid; i < CHUNK_ * DH_; i += 256) {
      const int r = i / DH_, d = i % DH_;
      const int p = c * CHUNK_ + r;
      const size_t qoff = (size_t)(b * S_ + p) * (H_ * DH_)   + h  * DH_ + d;
      const size_t koff = (size_t)(b * S_ + p) * (HKV_ * DH_) + hk * DH_ + d;
      const float ang = (float)p * ifreq_s[d & 63];
      const float cs = __cosf(ang), sn = __sinf(ang);
      float qo, ko;
      if (d < 64) {
        qo = bf2f(Q[qoff])  * cs - bf2f(Q[qoff + 64])  * sn;
        ko = bf2f(Kb[koff]) * cs - bf2f(Kb[koff + 64]) * sn;
      } else {
        qo = bf2f(Q[qoff])  * cs + bf2f(Q[qoff - 64])  * sn;
        ko = bf2f(Kb[koff]) * cs + bf2f(Kb[koff - 64]) * sn;
      }
      q_s[r][d] = f2bf(qo);
      k_s[r][d] = f2bf(ko);
    }
    wait_async0();
    __syncthreads();

    // ---- phase 1: scores = (q @ k^T) * decay-mask -> sc_s (bf16)
    for (int t = wave; t < 16; t += 8) {
      const int tr = (t >> 2) * 16, tc = (t & 3) * 16;
      v8f acc = {};
#pragma unroll
      for (int kk = 0; kk < DH_; kk += 32) {
        v16bf af = frag_row(&q_s[tr][0], DH_, lane, kk);
        v16bf bf = frag_row(&k_s[tc][0], DH_, lane, kk);
        acc = wmma_bf16(af, bf, acc);
      }
      const int j  = tc + (lane & 15);
      const int ib = tr + ((lane < 16) ? 0 : 8);
#pragma unroll
      for (int r = 0; r < 8; ++r) {
        const int i = ib + r;
        // branchless: unconditional table load (wrapped index) + value select
        const float w = dec_s[(i - j) & (CHUNK_ - 1)];
        sc_s[i][j] = f2bf((i >= j) ? acc[r] * w : 0.0f);
      }
    }
    __syncthreads();

    // ---- phase 2: in-place decay scaling (scores already consumed raw q,k)
    for (int i = tid; i < CHUNK_ * DH_; i += 256) {
      const int r = i / DH_, d = i % DH_;
      q_s[r][d] = f2bf(bf2f(q_s[r][d]) * qdec_s[r]);
      k_s[r][d] = f2bf(bf2f(k_s[r][d]) * kdec_s[r]);
    }
    __syncthreads();

    // ---- phase 3: o = scores @ v + (q*q_dec) @ kv   -> O (bf16)
    for (int t = wave; t < 32; t += 8) {
      const int tr = (t >> 3) * 16, tc = (t & 7) * 16;
      v8f acc = {};
#pragma unroll
      for (int kk = 0; kk < CHUNK_; kk += 32) {            // intra-chunk
        v16bf af = frag_row(&sc_s[tr][0], CHUNK_, lane, kk);
        v16bf bf = frag_col(&v_s[0][tc], DH_, lane, kk);
        acc = wmma_bf16(af, bf, acc);
      }
      const int n = lane & 15;
#pragma unroll
      for (int kk = 0; kk < DH_; kk += 32) {               // inter-chunk (kv state)
        v16bf af = frag_row(&q_s[tr][0], DH_, lane, kk);
        const int kb = kk + ((lane < 16) ? 0 : 8);
        v16bf bf;
#pragma unroll
        for (int u = 0; u < 8; ++u) {
          bf[u]     = f2bf(kv[(size_t)(kb + u) * DH_ + tc + n]);
          bf[u + 8] = f2bf(kv[(size_t)(kb + 16 + u) * DH_ + tc + n]);
        }
        acc = wmma_bf16(af, bf, acc);
      }
      const int n0 = tc + n;
      const int ib = tr + ((lane < 16) ? 0 : 8);
#pragma unroll
      for (int r = 0; r < 8; ++r)
        O[(size_t)(b * S_ + c * CHUNK_ + ib + r) * (H_ * DH_) + h * DH_ + n0] = f2bf(acc[r]);
    }
    __syncthreads();     // all kv reads complete before update

    // ---- phase 4: kv = blk * kv + (k*k_dec)^T @ v
    for (int t = wave; t < 64; t += 8) {
      const int tr = (t >> 3) * 16, tc = (t & 7) * 16;     // tr: d-index, tc: e-index
      v8f acc = {};
      const int d = tr + (lane & 15);
#pragma unroll
      for (int kk = 0; kk < CHUNK_; kk += 32) {
        const int jb = kk + ((lane < 16) ? 0 : 8);
        v16bf af;                                          // A(d,j) = k_s[j][d] (pre-scaled)
#pragma unroll
        for (int u = 0; u < 8; ++u) {
          af[u]     = k_s[jb + u][d];
          af[u + 8] = k_s[jb + 16 + u][d];
        }
        v16bf bf = frag_col(&v_s[0][tc], DH_, lane, kk);
        acc = wmma_bf16(af, bf, acc);
      }
      const int e0 = tc + (lane & 15);
      const int db = tr + ((lane < 16) ? 0 : 8);
#pragma unroll
      for (int r = 0; r < 8; ++r) {
        const size_t off = (size_t)(db + r) * DH_ + e0;
        kv[off] = blkdec * kv[off] + acc[r];
      }
    }
    __threadfence();     // kv writes visible to all waves next chunk
    __syncthreads();
  }
}

// ---------------------------------------------------------------------------
extern "C" void kernel_launch(void* const* d_in, const int* in_sizes, int n_in,
                              void* d_out, int out_size, void* d_ws, size_t ws_size,
                              hipStream_t stream) {
  (void)in_sizes; (void)n_in; (void)out_size; (void)ws_size;
  const float* x  = (const float*)d_in[0];
  const float* wq = (const float*)d_in[1];
  const float* wk = (const float*)d_in[2];
  const float* wv = (const float*)d_in[3];
  const float* wo = (const float*)d_in[4];
  float* out = (float*)d_out;

  // workspace carve-up (bf16 buffers first, then fp32 kv slab)
  __bf16* xb  = (__bf16*)d_ws;
  __bf16* wqb = xb  + (size_t)MROWS_ * D_;            // 8192*2048
  __bf16* wkb = wqb + (size_t)(H_ * DH_) * D_;        // 2048*2048
  __bf16* wvb = wkb + (size_t)(HKV_ * DH_) * D_;      //  512*2048
  __bf16* wob = wvb + (size_t)(HKV_ * DH_) * D_;      //  512*2048
  __bf16* Qb  = wob + (size_t)D_ * (H_ * DH_);        // 2048*2048
  __bf16* Kb  = Qb  + (size_t)MROWS_ * (H_ * DH_);
  __bf16* Vb  = Kb  + (size_t)MROWS_ * (HKV_ * DH_);
  __bf16* Ob  = Vb  + (size_t)MROWS_ * (HKV_ * DH_);
  float*  KV  = (float*)(Ob + (size_t)MROWS_ * (H_ * DH_));   // 32*128*128 fp32

  const dim3 blk(256);
  // one-shot fp32 -> bf16 casts
  cast_bf16_kernel<<<dim3(8192), blk, 0, stream>>>(x,  xb,  MROWS_ * D_);
  cast_bf16_kernel<<<dim3(4096), blk, 0, stream>>>(wq, wqb, (H_ * DH_) * D_);
  cast_bf16_kernel<<<dim3(1024), blk, 0, stream>>>(wk, wkb, (HKV_ * DH_) * D_);
  cast_bf16_kernel<<<dim3(1024), blk, 0, stream>>>(wv, wvb, (HKV_ * DH_) * D_);
  cast_bf16_kernel<<<dim3(4096), blk, 0, stream>>>(wo, wob, D_ * (H_ * DH_));

  // QKV projections (bf16 in, bf16 out)
  gemm_bf16_kernel<__bf16><<<dim3((H_ * DH_) / 64,   MROWS_ / 128), blk, 0, stream>>>(
      xb, wqb, Qb, MROWS_, H_ * DH_, D_);
  gemm_bf16_kernel<__bf16><<<dim3((HKV_ * DH_) / 64, MROWS_ / 128), blk, 0, stream>>>(
      xb, wkb, Kb, MROWS_, HKV_ * DH_, D_);
  gemm_bf16_kernel<__bf16><<<dim3((HKV_ * DH_) / 64, MROWS_ / 128), blk, 0, stream>>>(
      xb, wvb, Vb, MROWS_, HKV_ * DH_, D_);

  // chunked linear attention with fused RoPE
  lightning_attn_kernel<<<dim3(B_ * H_), blk, 0, stream>>>(Qb, Kb, Vb, Ob, KV);

  // output projection (bf16 in, fp32 out)
  gemm_bf16_kernel<float><<<dim3(D_ / 64, MROWS_ / 128), blk, 0, stream>>>(
      Ob, wob, out, MROWS_, D_, H_ * DH_);
}
